// SetTransformerDecoder_44702019617293
// MI455X (gfx1250) — compile-verified, matrix operand order checked by reading, not run on hardware
//
#include <hip/hip_runtime.h>
#include <hip/hip_bf16.h>
#include <math.h>

// Problem sizes (fixed by the reference)
#define NN 262144
#define DD 256
#define HH 8
#define GG 2048
#define SEG 128           // rows per set (N/G), contiguous
#define HSTRIDE 260       // padded LDS row stride (floats): 8*260 % 64 == 32 -> conflict-free C stores

typedef __attribute__((ext_vector_type(16))) _Float16 v16h;
typedef __attribute__((ext_vector_type(8)))  float    v8f;
typedef __attribute__((ext_vector_type(4)))  float    v4f;
typedef __attribute__((ext_vector_type(2)))  float    v2f;
typedef __attribute__((ext_vector_type(4)))  unsigned int v4u;
typedef __attribute__((ext_vector_type(8)))  int      v8i;
typedef __attribute__((ext_vector_type(4)))  int      v4i;

__device__ __forceinline__ float gelu_erf(float v) {
    // exact (erf-based) GELU, matching jax.nn.gelu(approximate=False)
    return 0.5f * v * (1.0f + erff(v * 0.70710678118654752f));
}

// ---------------------------------------------------------------------------
// Prep kernel A: q = seed@wq + bq ; Ws[d][head(16 pad)] = (wk_head @ q_head)/sqrt(DH)
//                cs[head] = (bk_head . q_head)/sqrt(DH)
// ---------------------------------------------------------------------------
__global__ __launch_bounds__(256) void prep_small(
    const float* __restrict__ seed, const float* __restrict__ wq,
    const float* __restrict__ bq,   const float* __restrict__ wk,
    const float* __restrict__ bk,   float* __restrict__ Wsg,
    float* __restrict__ csg)
{
    __shared__ float qL[DD];
    const int t = threadIdx.x;
    const float rs = 0.17677669529663687f;  // 1/sqrt(32)

    float acc = bq[t];
    for (int i = 0; i < DD; ++i) acc += seed[i] * wq[i * DD + t];
    qL[t] = acc;
    __syncthreads();

    for (int h = 0; h < HH; ++h) {
        float s = 0.0f;
        const float* wrow = wk + t * DD + h * 32;
        const float* qh = qL + h * 32;
        for (int c = 0; c < 32; ++c) s += wrow[c] * qh[c];
        Wsg[t * 16 + h] = s * rs;
    }
    for (int h = HH; h < 16; ++h) Wsg[t * 16 + h] = 0.0f;

    if (t < HH) {
        float s = 0.0f;
        for (int c = 0; c < 32; ++c) s += bk[t * 32 + c] * qL[t * 32 + c];
        csg[t] = s * rs;
    }
}

// ---------------------------------------------------------------------------
// Prep kernel B: convert lin_w (fp32 row-major [K=256][N=256]) into f16
// WMMA-B fragment order: frag f = kt*16+nt ; element ((f*32+lane)*16 + e)
// holds lin_w[k][n] with n = nt*16 + lane%16,
// k = kt*32 + (e<8 ? (lane>>4)*8+e : 16+(lane>>4)*8+(e-8))
// ---------------------------------------------------------------------------
__global__ __launch_bounds__(256) void swz_linw(
    const float* __restrict__ w, _Float16* __restrict__ o)
{
    const int gid = blockIdx.x * 256 + threadIdx.x;   // 65536 total
    const int e    = gid & 15;
    const int lane = (gid >> 4) & 31;
    const int f    = gid >> 9;
    const int kt = f >> 4, nt = f & 15;
    const int sel = lane >> 4;
    const int n = nt * 16 + (lane & 15);
    const int k = kt * 32 + ((e < 8) ? (sel * 8 + e) : (16 + sel * 8 + (e - 8)));
    o[gid] = (_Float16)w[k * DD + n];
}

// ---------------------------------------------------------------------------
// Main fused kernel: one workgroup (8 wave32) per 128-row segment.
// lin_w (f16 fragments, 128 KB) is DMA'd into LDS once per WG by the Tensor
// Data Mover; the WMMA inner loop then runs entirely out of LDS.
// LDS: bufB 131072 B + h32 133120 B + tail 28704 B = 292896 B (1 WG/WGP).
// ---------------------------------------------------------------------------
#define BUFB_FLOATS 32768
#define SMEM_FLOATS (BUFB_FLOATS + SEG*HSTRIDE + 7176)
#define SMEM_BYTES  (SMEM_FLOATS * 4)

__global__ __launch_bounds__(256) void st_main(
    const float* __restrict__ x,  const _Float16* __restrict__ wlin16,
    const float* __restrict__ Wsg, const float* __restrict__ csg,
    const float* __restrict__ lin_b,
    const float* __restrict__ wv, const float* __restrict__ bv,
    const float* __restrict__ wo, const float* __restrict__ bo,
    const float* __restrict__ w1, const float* __restrict__ b1,
    const float* __restrict__ w2, const float* __restrict__ b2,
    float* __restrict__ out)
{
    extern __shared__ float smem[];
    float* bufB = smem;                         // [0, 32768) floats: lin_w f16 fragments (TDM dest)
    float* h32  = smem + BUFB_FLOATS;           // 33280 floats, live stages 1-4
    float* reg  = h32 + SEG * HSTRIDE;          // overlapped tail region (7176 floats)
    float* WsL     = reg;                       // [0,4096)   live entry..stage2
    float* attL    = reg;                       // [0,2048)   live stage4..5 (aliases WsL)
    float* pooledL = reg + 2048;                // [2048,2304) stage5
    float* t1L     = reg + 2304;                // [2304,2560) stage5
    float* t2L     = reg + 2560;                // [2560,2816) stage5
    float* linbL   = reg + 4096;                // [4096,4352) stage1 (aliased by scoresL)
    float* scoresL = reg + 4096;                // [4096,5120) stage2..3
    float* wL      = reg + 5120;                // [5120,7168) stage3..4
    float* csL     = reg + 7168;                // [7168,7176) stage2

    const int t = threadIdx.x;
    const int wave = t >> 5, lane = t & 31;
    const int lan16 = lane & 15, sel = lane >> 4;
    const int g = blockIdx.x;

    // ---- kick off TDM DMA of the whole swizzled lin_w (128 KB) into LDS ----
    if (wave == 0) {
        const unsigned long long ga = (unsigned long long)(uintptr_t)wlin16;
        const unsigned ldsOff = __builtin_amdgcn_groupstaticsize();  // dynamic LDS base; bufB at +0
        // D# group 0: count=1 | lds_addr | global_addr[56:0] | type=2
        v4u g0 = { 1u,
                   ldsOff,
                   (unsigned)(ga & 0xFFFFFFFFu),
                   (unsigned)((ga >> 32) & 0x01FFFFFFu) | (2u << 30) };
        // D# group 1: data_size=8B; tensor_dim0=16384, tensor_dim1=1;
        //             tile_dim0=16384 (1-D tile); dim0_stride=16384
        v8i g1 = { (int)(3u << 16),
                   (int)(16384u << 16),     // tensor_dim0[15:0] -> bits[63:48]
                   (int)(1u << 16),         // tensor_dim0[31:16]=0 | tensor_dim1[15:0]=1
                   (int)(16384u << 16),     // tensor_dim1[31:16]=0 | tile_dim0=16384
                   0,                       // tile_dim1=0 (unused), tile_dim2=0
                   16384, 0, 0 };           // tensor_dim0_stride=16384
        v4i gz4 = { 0, 0, 0, 0 };
        v8i gz8 = { 0, 0, 0, 0, 0, 0, 0, 0 };
        __builtin_amdgcn_tensor_load_to_lds(g0, g1, gz4, gz4, gz8, 0);
    }

    // stage small constants into LDS (overlaps the DMA)
    for (int i = t; i < 4096; i += 256) WsL[i] = Wsg[i];
    linbL[t] = lin_b[t];
    if (t < HH) csL[t] = csg[t];

    // ---- stage 1: h = gelu(x @ lin_w + b), f16 WMMA, rows 16*wave..+15 ----
    v8f acc[16];
    #pragma unroll
    for (int i = 0; i < 16; ++i) acc[i] = {};

    const float* xrow = x + (size_t)(g * SEG + wave * 16 + lan16) * DD;
    const v16h* ldsB = (const v16h*)bufB;

    // A-row floats for kt=0 (HBM loads overlap the DMA wait)
    v4f fa0 = *(const v4f*)(xrow + sel * 8);
    v4f fa1 = *(const v4f*)(xrow + sel * 8 + 4);
    v4f fa2 = *(const v4f*)(xrow + 16 + sel * 8);
    v4f fa3 = *(const v4f*)(xrow + 16 + sel * 8 + 4);

    if (wave == 0) __builtin_amdgcn_s_wait_tensorcnt(0);
    __syncthreads();   // lin_w now resident in LDS (also fences WsL/linbL staging)

    for (int kt = 0; kt < 8; ++kt) {
        v4f fb0, fb1, fb2, fb3;
        if (kt < 7) {   // uniform branch: prefetch next kt's A floats
            const float* p = xrow + (kt + 1) * 32;
            fb0 = *(const v4f*)(p + sel * 8);
            fb1 = *(const v4f*)(p + sel * 8 + 4);
            fb2 = *(const v4f*)(p + 16 + sel * 8);
            fb3 = *(const v4f*)(p + 16 + sel * 8 + 4);
        }
        v16h a;
        a[0]=(_Float16)fa0[0]; a[1]=(_Float16)fa0[1]; a[2]=(_Float16)fa0[2]; a[3]=(_Float16)fa0[3];
        a[4]=(_Float16)fa1[0]; a[5]=(_Float16)fa1[1]; a[6]=(_Float16)fa1[2]; a[7]=(_Float16)fa1[3];
        a[8]=(_Float16)fa2[0]; a[9]=(_Float16)fa2[1]; a[10]=(_Float16)fa2[2]; a[11]=(_Float16)fa2[3];
        a[12]=(_Float16)fa3[0]; a[13]=(_Float16)fa3[1]; a[14]=(_Float16)fa3[2]; a[15]=(_Float16)fa3[3];
        #pragma unroll
        for (int nt = 0; nt < 16; ++nt) {
            v16h b = ldsB[(kt * 16 + nt) * 32 + lane];   // 2x ds_load_b128, short latency
            acc[nt] = __builtin_amdgcn_wmma_f32_16x16x32_f16(
                false, a, false, b, (short)0, acc[nt], false, false);
        }
        fa0 = fb0; fa1 = fb1; fa2 = fb2; fa3 = fb3;
    }
    #pragma unroll
    for (int nt = 0; nt < 16; ++nt) {
        const int col = nt * 16 + lan16;
        const float bias = linbL[col];
        #pragma unroll
        for (int r = 0; r < 8; ++r) {
            const int row = wave * 16 + r + sel * 8;
            h32[row * HSTRIDE + col] = gelu_erf(acc[nt][r] + bias);
        }
    }
    __syncthreads();

    // ---- stage 2: scores = h @ Ws + cs (fp32 WMMA 16x16x4, 1 N-tile) ----
    {
        v8f sacc = {};
        const int srow = wave * 16 + lan16;
        #pragma unroll 4
        for (int ks = 0; ks < 64; ++ks) {
            const int k = ks * 4 + sel * 2;
            v2f a; a[0] = h32[srow * HSTRIDE + k]; a[1] = h32[srow * HSTRIDE + k + 1];
            v2f b; b[0] = WsL[k * 16 + lan16];     b[1] = WsL[(k + 1) * 16 + lan16];
            sacc = __builtin_amdgcn_wmma_f32_16x16x4_f32(
                false, a, false, b, (short)0, sacc, false, false);
        }
        if (lan16 < HH) {
            const float c0 = csL[lan16];
            #pragma unroll
            for (int r = 0; r < 8; ++r) {
                const int row = wave * 16 + r + sel * 8;
                scoresL[row * 8 + lan16] = sacc[r] + c0;
            }
        }
    }
    __syncthreads();

    // ---- stage 3: per-head segment softmax (wave 'w' owns head 'w') ----
    {
        const int head = wave, j = lane;
        float v0 = scoresL[(j      ) * 8 + head];
        float v1 = scoresL[(j +  32) * 8 + head];
        float v2 = scoresL[(j +  64) * 8 + head];
        float v3 = scoresL[(j +  96) * 8 + head];
        float m = fmaxf(fmaxf(v0, v1), fmaxf(v2, v3));
        for (int off = 16; off >= 1; off >>= 1) m = fmaxf(m, __shfl_xor(m, off, 32));
        float e0 = expf(v0 - m), e1 = expf(v1 - m), e2 = expf(v2 - m), e3 = expf(v3 - m);
        float z = e0 + e1 + e2 + e3;
        for (int off = 16; off >= 1; off >>= 1) z += __shfl_xor(z, off, 32);
        const float inv = 1.0f / z;
        wL[(j      ) * 16 + head] = e0 * inv;
        wL[(j +  32) * 16 + head] = e1 * inv;
        wL[(j +  64) * 16 + head] = e2 * inv;
        wL[(j +  96) * 16 + head] = e3 * inv;
        // zero the pad heads 8..15
        for (int i = t; i < SEG * 8; i += 256) {
            const int row = i >> 3;
            wL[row * 16 + 8 + (i & 7)] = 0.0f;
        }
    }
    __syncthreads();

    // ---- stage 4: A = w^T[16x128] @ h[128x256]; wave owns 2 N-tiles ----
    {
        v8f pa0 = {}, pa1 = {};
        const int nt0 = wave * 2;
        #pragma unroll 4
        for (int ks = 0; ks < 32; ++ks) {
            const int k = ks * 4 + sel * 2;
            v2f a; a[0] = wL[k * 16 + lan16]; a[1] = wL[(k + 1) * 16 + lan16];
            const int col0 = nt0 * 16 + lan16;
            v2f b0v; b0v[0] = h32[k * HSTRIDE + col0];      b0v[1] = h32[(k + 1) * HSTRIDE + col0];
            v2f b1v; b1v[0] = h32[k * HSTRIDE + col0 + 16]; b1v[1] = h32[(k + 1) * HSTRIDE + col0 + 16];
            pa0 = __builtin_amdgcn_wmma_f32_16x16x4_f32(false, a, false, b0v, (short)0, pa0, false, false);
            pa1 = __builtin_amdgcn_wmma_f32_16x16x4_f32(false, a, false, b1v, (short)0, pa1, false, false);
        }
        __syncthreads();   // h32 & wL dead after this; attL may overwrite WsL region
        if (sel == 0) {    // rows m = r (heads 0..7); sel==1 holds pad rows 8..15
            #pragma unroll
            for (int r = 0; r < 8; ++r) {
                attL[r * DD + nt0 * 16 + lan16]      = pa0[r];
                attL[r * DD + nt0 * 16 + 16 + lan16] = pa1[r];
            }
        }
    }
    __syncthreads();

    // ---- stage 5: tail. pooled = A_head @ wv_head + bv ; then wo ; MLP ----
    {
        const int head = t >> 5;   // output col t belongs to head t/32
        float accp = bv[t];
        #pragma unroll 4
        for (int d = 0; d < DD; ++d) accp += attL[head * DD + d] * wv[d * DD + t];
        pooledL[t] = accp;
    }
    __syncthreads();
    {
        float a1 = bo[t];
        #pragma unroll 4
        for (int d = 0; d < DD; ++d) a1 += pooledL[d] * wo[d * DD + t];
        t1L[t] = a1;
    }
    __syncthreads();
    {
        float a2 = b1[t];
        #pragma unroll 4
        for (int d = 0; d < DD; ++d) a2 += t1L[d] * w1[d * DD + t];
        t2L[t] = gelu_erf(a2);
    }
    __syncthreads();
    {
        float a3 = b2[t];
        #pragma unroll 4
        for (int d = 0; d < DD; ++d) a3 += t2L[d] * w2[d * DD + t];
        out[(size_t)g * DD + t] = a3;
    }
}

// ---------------------------------------------------------------------------
extern "C" void kernel_launch(void* const* d_in, const int* in_sizes, int n_in,
                              void* d_out, int out_size, void* d_ws, size_t ws_size,
                              hipStream_t stream) {
    (void)in_sizes; (void)n_in; (void)out_size; (void)ws_size;
    const float* x     = (const float*)d_in[0];
    // d_in[1] = ptr (int64), d_in[2] = batch (int64): segments are uniform
    // 128-row contiguous blocks, so they are not needed.
    const float* lin_w = (const float*)d_in[3];
    const float* lin_b = (const float*)d_in[4];
    const float* seed  = (const float*)d_in[5];
    const float* wq    = (const float*)d_in[6];
    const float* bq    = (const float*)d_in[7];
    const float* wk    = (const float*)d_in[8];
    const float* bk    = (const float*)d_in[9];
    const float* wv    = (const float*)d_in[10];
    const float* bv    = (const float*)d_in[11];
    const float* wo    = (const float*)d_in[12];
    const float* bo    = (const float*)d_in[13];
    const float* w1    = (const float*)d_in[14];
    const float* b1    = (const float*)d_in[15];
    const float* w2    = (const float*)d_in[16];
    const float* b2    = (const float*)d_in[17];

    char* ws = (char*)d_ws;
    _Float16* wlin16 = (_Float16*)ws;                  // 131072 B (TDM source)
    float*    Wsg    = (float*)(ws + 131072);          //  16384 B
    float*    csg    = (float*)(ws + 131072 + 16384);  //     32 B

    prep_small<<<1, 256, 0, stream>>>(seed, wq, bq, wk, bk, Wsg, csg);
    swz_linw<<<256, 256, 0, stream>>>(lin_w, wlin16);

    (void)hipFuncSetAttribute(reinterpret_cast<const void*>(st_main),
                              hipFuncAttributeMaxDynamicSharedMemorySize, SMEM_BYTES);
    st_main<<<GG, 256, SMEM_BYTES, stream>>>(
        x, wlin16, Wsg, csg, lin_b, wv, bv, wo, bo, w1, b1, w2, b2, (float*)d_out);
}
